// GnnPolicy_27487790695248
// MI455X (gfx1250) — matrix operand order repaired and checked by reading, not compiled, not run on hardware
//
#include <hip/hip_runtime.h>
#include <hip/hip_bf16.h>

// ---------------------------------------------------------------------------
// GNN policy: encoder MLP -> 3x (pairwise msg MLP + sum + update MLP)
// B=32 N=128 OBS=32 H=M=128 L=3.  All GEMMs done with v_wmma_f32_16x16x32_f16.
// ---------------------------------------------------------------------------

typedef _Float16 h4   __attribute__((ext_vector_type(4)));
typedef _Float16 h8   __attribute__((ext_vector_type(8)));
typedef _Float16 v16h __attribute__((ext_vector_type(16)));
typedef float    v8f  __attribute__((ext_vector_type(8)));

#define BN 4096   // B*N rows
#define HD 128    // hidden width (N dim of every GEMM)

__device__ __forceinline__ float n2nf(float x, float pinf = 1000.f, float ninf = -1000.f) {
  if (__builtin_isnan(x)) return 0.f;
  if (__builtin_isinf(x)) return x > 0.f ? pinf : ninf;
  return x;
}

__device__ __forceinline__ v8f wmma16(v16h a, v16h b, v8f c) {
  // D = A(16x32 f16) * B(32x16 f16) + C(16x16 f32)
  return __builtin_amdgcn_wmma_f32_16x16x32_f16(false, a, false, b, (short)0, c, false, false);
}

// A fragment: row-major [16][K] f16 panel in LDS.
// lane 0-15: halves 0-7 -> K=kt*32+0..7,  halves 8-15 -> K=kt*32+16..23
// lane16-31: halves 0-7 -> K=kt*32+8..15, halves 8-15 -> K=kt*32+24..31
__device__ __forceinline__ v16h frag_a(const _Float16* p, int K, int mrow, int kt, int lane) {
  int off = kt * 32 + ((lane & 16) ? 8 : 0);
  union { v16h v; h8 h[2]; } u;
  u.h[0] = *(const h8*)(p + mrow * K + off);
  u.h[1] = *(const h8*)(p + mrow * K + off + 16);
  return u.v;
}
// B fragment from W^T panel [N][K] f16 in LDS.
// lane 0-15: col=lane,    K = kt*32+0..7,16..23 ; lane16-31: col=lane-16, K = kt*32+8..15,24..31
__device__ __forceinline__ v16h frag_b(const _Float16* p, int K, int ncol, int kt, int lane) {
  int off = kt * 32 + ((lane & 16) ? 8 : 0);
  union { v16h v; h8 h[2]; } u;
  u.h[0] = *(const h8*)(p + ncol * K + off);
  u.h[1] = *(const h8*)(p + ncol * K + off + 16);
  return u.v;
}

// One 16-row strip x 128 cols of WMMA work over a [128][K] A panel and a
// [128][K] W^T panel, both f16 in LDS.  Per k-step we force the pre-RA
// scheduler to batch all 18 ds_reads (2 A + 16 B) ahead of the 8 WMMAs via
// sched_group_barrier; this keeps the 8 B fragments in distinct registers so
// the hardware waits become partial (s_wait_dscnt N) instead of a full
// dscnt==0 drain before every WMMA.
template <int K>
__device__ __forceinline__ void gemm_strip(const _Float16* lA, const _Float16* lB,
                                           int mrow, int colBase, int lane, v8f acc[8]) {
#pragma unroll
  for (int kt = 0; kt < K / 32; ++kt) {
    v16h a = frag_a(lA, K, mrow, kt, lane);
    v16h b[8];
#pragma unroll
    for (int t = 0; t < 8; ++t)
      b[t] = frag_b(lB, K, t * 16 + colBase, kt, lane);
#pragma unroll
    for (int t = 0; t < 8; ++t)
      acc[t] = wmma16(a, b[t], acc[t]);
    // 0x100 = DS read, 0x008 = MFMA/WMMA
    __builtin_amdgcn_sched_group_barrier(0x100, 18, 0);
    __builtin_amdgcn_sched_group_barrier(0x008, 8, 0);
  }
}

// ---------------------------------------------------------------------------
// Generic tiled GEMM: Y(M x 128) = act(X(M x K, f32) @ Wt^T + bias)
// Wt is stored transposed (128 x K) in f16. Block = 256 threads = 8 waves,
// computes a 128x128 output tile. flags: 1=relu, 2=nan_to_num, 4=sanitize A.
// ---------------------------------------------------------------------------
template <int K>
__global__ __launch_bounds__(256) void wgemm128(
    const float* __restrict__ X, const _Float16* __restrict__ Wt,
    const float* __restrict__ bias, float* __restrict__ Y, int flags) {
  extern __shared__ _Float16 sm[];
  _Float16* lA = sm;                 // [128][K]
  _Float16* lB = sm + 128 * K;       // [128][K]  (W^T)
  const int tid  = threadIdx.x;
  const int lane = tid & 31, wave = tid >> 5;
  const int rowBlk = blockIdx.x * 128;
  const int elems = 128 * K;

  // load+convert A panel (rows are contiguous since ld == K)
  const float* Xb = X + (size_t)rowBlk * K;
#pragma unroll
  for (int idx = 0; idx < elems / (256 * 4); ++idx) {
    int o = idx * 256 * 4 + tid * 4;
    float4 v = *(const float4*)(Xb + o);
    if (flags & 4) { v.x = n2nf(v.x); v.y = n2nf(v.y); v.z = n2nf(v.z); v.w = n2nf(v.w); }
    h4 hv = { (_Float16)v.x, (_Float16)v.y, (_Float16)v.z, (_Float16)v.w };
    *(h4*)(lA + o) = hv;
  }
  // load W^T panel (already f16, contiguous)
#pragma unroll
  for (int idx = 0; idx < elems / (256 * 8); ++idx) {
    int o = idx * 256 * 8 + tid * 8;
    *(h8*)(lB + o) = *(const h8*)(Wt + o);
  }
  __syncthreads();

  v8f acc[8] = {};
  const int colBase = lane & 15;
  const int mrow = wave * 16 + colBase;
  gemm_strip<K>(lA, lB, mrow, colBase, lane, acc);

  const int rbase = rowBlk + wave * 16 + ((lane >> 4) << 3);
#pragma unroll
  for (int t = 0; t < 8; ++t) {
    int col = t * 16 + colBase;
    float bb = bias ? bias[col] : 0.f;
#pragma unroll
    for (int r = 0; r < 8; ++r) {
      float v = acc[t][r] + bb;
      if (flags & 1) v = fmaxf(v, 0.f);
      if (flags & 2) v = n2nf(v);
      Y[(size_t)(rbase + r) * HD + col] = v;
    }
  }
}

// ---------------------------------------------------------------------------
// Heavy message kernel: one block per (b,i).
//   h1[j,:] = relu(C[b,i,:] + G[b,j,:] + d_ij * w1d[:])       (b1 folded in G)
//   h2 = relu(h1 @ W2 + b2);  t = h2 @ W3 + b3; t[i,:] = 0
//   msg[b,i,:] = sum_j t[j,:]
// ---------------------------------------------------------------------------
__global__ __launch_bounds__(256) void msg_kernel(
    const float* __restrict__ Cbuf, const float* __restrict__ Gbuf,
    const float* __restrict__ dij,  const float* __restrict__ w1d,
    const _Float16* __restrict__ W2t, const _Float16* __restrict__ W3t,
    const float* __restrict__ b2, const float* __restrict__ b3,
    float* __restrict__ msg) {
  extern __shared__ _Float16 sm[];
  _Float16* lH1 = sm;                  // [128][128] f16
  _Float16* lH2 = sm + 16384;
  _Float16* lW2 = sm + 32768;
  _Float16* lW3 = sm + 49152;
  float*    lmsg = (float*)(sm + 65536);  // [128] f32

  const int tid = threadIdx.x, lane = tid & 31, wave = tid >> 5;
  const int bi = blockIdx.x;           // b*128 + i
  const int b = bi >> 7, i = bi & 127;

  if (tid < HD) lmsg[tid] = 0.f;

#pragma unroll
  for (int idx = 0; idx < 16384 / (256 * 8); ++idx) {
    int o = idx * 256 * 8 + tid * 8;
    *(h8*)(lW2 + o) = *(const h8*)(W2t + o);
    *(h8*)(lW3 + o) = *(const h8*)(W3t + o);
  }

  const float* Ci = Cbuf + (size_t)bi * HD;
  const float* Gb = Gbuf + (size_t)b * 128 * HD;
  const float* dv = dij + (size_t)bi * 128;
#pragma unroll
  for (int idx = 0; idx < 16384 / (256 * 4); ++idx) {
    int o = idx * 256 * 4 + tid * 4;
    int j = o >> 7, c = o & 127;
    float4 g  = *(const float4*)(Gb + j * HD + c);
    float4 ci = *(const float4*)(Ci + c);
    float4 wd = *(const float4*)(w1d + c);
    float d = dv[j];
    h4 hv = { (_Float16)fmaxf(ci.x + g.x + d * wd.x, 0.f),
              (_Float16)fmaxf(ci.y + g.y + d * wd.y, 0.f),
              (_Float16)fmaxf(ci.z + g.z + d * wd.z, 0.f),
              (_Float16)fmaxf(ci.w + g.w + d * wd.w, 0.f) };
    *(h4*)(lH1 + o) = hv;
  }
  __syncthreads();

  const int colBase = lane & 15;
  const int mrow = wave * 16 + colBase;
  const int rbase = wave * 16 + ((lane >> 4) << 3);

  { // h2 = relu(h1 @ W2 + b2) -> lH2 (f16)
    v8f acc[8] = {};
    gemm_strip<HD>(lH1, lW2, mrow, colBase, lane, acc);
#pragma unroll
    for (int t = 0; t < 8; ++t) {
      int col = t * 16 + colBase;
      float bb = b2[col];
#pragma unroll
      for (int r = 0; r < 8; ++r)
        lH2[(rbase + r) * HD + col] = (_Float16)fmaxf(acc[t][r] + bb, 0.f);
    }
  }
  __syncthreads();

  { // t = h2 @ W3 + b3; mask j==i; reduce over j into lmsg
    v8f acc[8] = {};
    gemm_strip<HD>(lH2, lW3, mrow, colBase, lane, acc);
#pragma unroll
    for (int t = 0; t < 8; ++t) {
      int col = t * 16 + colBase;
      float bb = b3[col];
      float s = 0.f;
#pragma unroll
      for (int r = 0; r < 8; ++r) {
        int j = rbase + r;
        float v = acc[t][r] + bb;
        s += (j == i) ? 0.f : v;
      }
      atomicAdd(&lmsg[col], s);
    }
  }
  __syncthreads();
  if (tid < HD) msg[(size_t)bi * HD + tid] = lmsg[tid];
}

// ---------------------------------------------------------------------------
// Small helper kernels
// ---------------------------------------------------------------------------
// W (K x 128, f32, ld=128) -> Wt (128 x K, f16)
__global__ void tconv_kernel(const float* __restrict__ W, _Float16* __restrict__ Wt, int K) {
  int idx = blockIdx.x * 256 + threadIdx.x;
  if (idx >= 128 * K) return;
  int n = idx / K, k = idx - n * K;
  Wt[idx] = (_Float16)W[k * 128 + n];
}

__global__ void dij_kernel(const float* __restrict__ pos, float* __restrict__ dij) {
  int idx = blockIdx.x * 256 + threadIdx.x;     // over B*N*N = 524288
  int b = idx >> 14, ij = idx & 16383;
  int i = ij >> 7, j = ij & 127;
  float xi = n2nf(pos[(b * 128 + i) * 2 + 0]);
  float yi = n2nf(pos[(b * 128 + i) * 2 + 1]);
  float xj = n2nf(pos[(b * 128 + j) * 2 + 0]);
  float yj = n2nf(pos[(b * 128 + j) * 2 + 1]);
  float dx = xi - xj, dy = yi - yj;
  float d2 = dx * dx + dy * dy;
  float d = (d2 > 0.f) ? sqrtf(d2) : 0.f;
  if (__builtin_isnan(d)) d = 0.f;
  else if (__builtin_isinf(d)) d = 1000.f;
  dij[idx] = d;
}

__global__ void concat_kernel(const float* __restrict__ z, const float* __restrict__ m,
                              float* __restrict__ cat) {
  int idx = blockIdx.x * 256 + threadIdx.x;     // over BN*128
  int row = idx >> 7, c = idx & 127;
  cat[(size_t)row * 256 + c] = z[idx];
  cat[(size_t)row * 256 + 128 + c] = m[idx];
}

// ---------------------------------------------------------------------------
extern "C" void kernel_launch(void* const* d_in, const int* in_sizes, int n_in,
                              void* d_out, int out_size, void* d_ws, size_t ws_size,
                              hipStream_t stream) {
  const float* obs    = (const float*)d_in[0];
  const float* pos    = (const float*)d_in[1];
  const float* enc_w1 = (const float*)d_in[2];
  const float* enc_b1 = (const float*)d_in[3];
  const float* enc_w2 = (const float*)d_in[4];
  const float* enc_b2 = (const float*)d_in[5];
  const float* enc_w3 = (const float*)d_in[6];
  const float* enc_b3 = (const float*)d_in[7];
  const float* msg_w1 = (const float*)d_in[8];
  const float* msg_b1 = (const float*)d_in[9];
  const float* msg_w2 = (const float*)d_in[10];
  const float* msg_b2 = (const float*)d_in[11];
  const float* msg_w3 = (const float*)d_in[12];
  const float* msg_b3 = (const float*)d_in[13];
  const float* upd_w1 = (const float*)d_in[14];
  const float* upd_b1 = (const float*)d_in[15];
  const float* upd_w2 = (const float*)d_in[16];
  const float* upd_b2 = (const float*)d_in[17];
  const float* upd_w3 = (const float*)d_in[18];
  const float* upd_b3 = (const float*)d_in[19];
  (void)in_sizes; (void)n_in; (void)out_size; (void)ws_size;

  char* ws = (char*)d_ws;
  size_t off = 0;
  auto alloc = [&](size_t bytes) -> char* {
    char* p = ws + off; off += (bytes + 255) & ~(size_t)255; return p;
  };

  float* dij  = (float*)alloc((size_t)BN * 128 * 4);
  float* zA   = (float*)alloc((size_t)BN * HD * 4);
  float* zB   = (float*)alloc((size_t)BN * HD * 4);
  float* Cb   = (float*)alloc((size_t)BN * HD * 4);   // also reused as MLP temp
  float* Gb   = (float*)alloc((size_t)BN * HD * 4);   // also reused as MLP temp
  float* msgb = (float*)alloc((size_t)BN * HD * 4);
  float* cat  = (float*)alloc((size_t)BN * 256 * 4);

  _Float16* E1t = (_Float16*)alloc(128 * 32 * 2);
  _Float16* E2t = (_Float16*)alloc(128 * 128 * 2);
  _Float16* E3t = (_Float16*)alloc(128 * 128 * 2);
  _Float16 *W1aT[3], *W1bT[3], *W2t[3], *W3t[3], *U1t[3], *U2t[3], *U3t[3];
  for (int l = 0; l < 3; ++l) {
    W1aT[l] = (_Float16*)alloc(128 * 128 * 2);
    W1bT[l] = (_Float16*)alloc(128 * 128 * 2);
    W2t[l]  = (_Float16*)alloc(128 * 128 * 2);
    W3t[l]  = (_Float16*)alloc(128 * 128 * 2);
    U1t[l]  = (_Float16*)alloc(128 * 256 * 2);
    U2t[l]  = (_Float16*)alloc(128 * 128 * 2);
    U3t[l]  = (_Float16*)alloc(128 * 128 * 2);
  }

  auto tconv = [&](const float* W, _Float16* Wt, int K) {
    int n = 128 * K;
    tconv_kernel<<<(n + 255) / 256, 256, 0, stream>>>(W, Wt, K);
  };
  tconv(enc_w1, E1t, 32);
  tconv(enc_w2, E2t, 128);
  tconv(enc_w3, E3t, 128);
  for (int l = 0; l < 3; ++l) {
    tconv(msg_w1 + (size_t)l * 257 * 128,              W1aT[l], 128);
    tconv(msg_w1 + (size_t)l * 257 * 128 + 128 * 128,  W1bT[l], 128);
    tconv(msg_w2 + (size_t)l * 128 * 128,              W2t[l],  128);
    tconv(msg_w3 + (size_t)l * 128 * 128,              W3t[l],  128);
    tconv(upd_w1 + (size_t)l * 256 * 128,              U1t[l],  256);
    tconv(upd_w2 + (size_t)l * 128 * 128,              U2t[l],  128);
    tconv(upd_w3 + (size_t)l * 128 * 128,              U3t[l],  128);
  }

  dij_kernel<<<(BN * 128) / 256, 256, 0, stream>>>(pos, dij);

  auto gemm32 = [&](const float* X, const _Float16* Wt, const float* bias,
                    float* Y, int flags) {
    wgemm128<32><<<BN / 128, 256, (size_t)2 * 128 * 32 * 2, stream>>>(X, Wt, bias, Y, flags);
  };
  auto gemm128 = [&](const float* X, const _Float16* Wt, const float* bias,
                     float* Y, int flags) {
    wgemm128<128><<<BN / 128, 256, (size_t)2 * 128 * 128 * 2, stream>>>(X, Wt, bias, Y, flags);
  };
  auto gemm256 = [&](const float* X, const _Float16* Wt, const float* bias,
                     float* Y, int flags) {
    wgemm128<256><<<BN / 128, 256, (size_t)2 * 128 * 256 * 2, stream>>>(X, Wt, bias, Y, flags);
  };

  // encoder: z = n2n(mlp(n2n(obs)))
  gemm32(obs, E1t, enc_b1, Cb, 1 | 4);     // relu + sanitize input
  gemm128(Cb, E2t, enc_b2, Gb, 1);         // relu
  gemm128(Gb, E3t, enc_b3, zA, 2);         // n2n

  float* zcur = zA;
  for (int l = 0; l < 3; ++l) {
    // C = z@W1a ; G = z@W1b + b1
    gemm128(zcur, W1aT[l], nullptr,                 Cb, 0);
    gemm128(zcur, W1bT[l], msg_b1 + (size_t)l * HD, Gb, 0);
    const float* w1d = msg_w1 + (size_t)l * 257 * 128 + 256 * 128;
    msg_kernel<<<BN, 256, 4 * 128 * 128 * 2 + 512, stream>>>(
        Cb, Gb, dij, w1d, W2t[l], W3t[l],
        msg_b2 + (size_t)l * HD, msg_b3 + (size_t)l * HD, msgb);
    // upd: z = n2n(mlp([z, msg]))
    concat_kernel<<<(BN * 128) / 256, 256, 0, stream>>>(zcur, msgb, cat);
    float* znext = (l == 2) ? (float*)d_out : ((zcur == zA) ? zB : zA);
    gemm256(cat, U1t[l], upd_b1 + (size_t)l * HD, Cb, 1);
    gemm128(Cb, U2t[l], upd_b2 + (size_t)l * HD, Gb, 1);
    gemm128(Gb, U3t[l], upd_b3 + (size_t)l * HD, znext, 2);
    zcur = znext;
  }
}
